// MultiLatentAttention_35304631173152
// MI455X (gfx1250) — matrix operand order repaired and checked by reading, not compiled, not run on hardware
//
#include <hip/hip_runtime.h>

typedef __bf16 bf16_t;
typedef __attribute__((ext_vector_type(16))) __bf16 v16bf;
typedef __attribute__((ext_vector_type(8)))  float  v8f;

#define T_SEQ   2048
#define DIM_    2048
#define H_HEADS 16
#define KV_RANK_ 512
#define Q_RANK_  1536
#define ROPE_    64
#define NOPE_    128
#define VD_      128
#define QKH_     192   // NOPE + ROPE

// ---------------------------------------------------------------- helpers
__device__ __forceinline__ unsigned mla_lds_off(const void* p) {
  // generic -> LDS(as3) pointer -> 32-bit LDS byte offset
  return (unsigned)(unsigned long long)(__attribute__((address_space(3))) const char*)(const char*)p;
}

// One async b128 instruction moves 32 lanes x 16B = 512B global -> LDS.
// Tile is ROWS x 32 bf16 (64B per row), LDS row stride = (32+8) bf16.
template<int ROWS>
__device__ __forceinline__ void mla_async_tile(const bf16_t* gbase, int ldg,
                                               bf16_t* lds_base, int wave, int lane) {
  #pragma unroll
  for (int j = 0; j < ROWS / 64; ++j) {
    int row = wave * (ROWS / 8) + j * 8 + (lane >> 2);
    const bf16_t* g = gbase + (size_t)row * ldg + (lane & 3) * 8;
    unsigned loff = mla_lds_off(lds_base + row * 40 + (lane & 3) * 8);
    asm volatile("global_load_async_to_lds_b128 %0, %1, off"
                 :: "v"(loff), "v"((unsigned long long)(size_t)g)
                 : "memory");
  }
}

// ---------------------------------------------------------------- cast f32 -> bf16
__global__ void mla_cast_f32_bf16(const float* __restrict__ src,
                                  bf16_t* __restrict__ dst, int n) {
  int i = blockIdx.x * blockDim.x + threadIdx.x;
  if (i < n) dst[i] = (bf16_t)src[i];
}

// ---------------------------------------------------------------- WMMA GEMM (async double-buffered)
// C[M,N] (f32) = A[M,K] (bf16, row major) * W[N,K]^T (bf16, row major)
// Requires M%128==0, N%BN==0, K%32==0.
template<int BN>
__global__ __launch_bounds__(256)
void mla_gemm_bf16_wmma(const bf16_t* __restrict__ A, const bf16_t* __restrict__ W,
                        float* __restrict__ C, int M, int N, int K) {
  constexpr int BM = 128, BK = 32, LDT = BK + 8;
  constexpr int WN = BN / 2;           // per-wave n extent
  constexpr int NT = WN / 16;          // n tiles per wave (2 or 4)
  constexpr int NLD = BM / 64 + BN / 64; // async load instrs per wave per chunk
  __shared__ bf16_t As[2][BM][LDT];
  __shared__ bf16_t Ws[2][BN][LDT];

  const int tid  = threadIdx.x;
  const int lane = tid & 31;
  const int wave = tid >> 5;     // 0..7
  const int wm   = wave >> 1;    // 0..3 : 32-row strip
  const int wn   = wave & 1;     // 0..1 : WN-col strip
  const int hl   = lane >> 4;
  const int l15  = lane & 15;
  const int bm0  = blockIdx.y * BM;
  const int bn0  = blockIdx.x * BN;

  v8f acc[2][NT];
  #pragma unroll
  for (int mt = 0; mt < 2; ++mt)
    #pragma unroll
    for (int nt = 0; nt < NT; ++nt)
      #pragma unroll
      for (int i = 0; i < 8; ++i) acc[mt][nt][i] = 0.f;

  const int nk = K / BK;
  // prologue: stream chunk 0 into buffer 0
  mla_async_tile<BM>(A + (size_t)bm0 * K, K, &As[0][0][0], wave, lane);
  mla_async_tile<BN>(W + (size_t)bn0 * K, K, &Ws[0][0][0], wave, lane);

  for (int kc = 0; kc < nk; ++kc) {
    const int cur = kc & 1;
    if (kc + 1 < nk) {  // stream next chunk into the other buffer
      const int k1 = (kc + 1) * BK;
      mla_async_tile<BM>(A + (size_t)bm0 * K + k1, K, &As[1 - cur][0][0], wave, lane);
      mla_async_tile<BN>(W + (size_t)bn0 * K + k1, K, &Ws[1 - cur][0][0], wave, lane);
      asm volatile("s_wait_asynccnt %0" :: "i"(NLD) : "memory"); // chunk kc done
    } else {
      asm volatile("s_wait_asynccnt 0" ::: "memory");
    }
    __syncthreads();   // all waves' chunk-kc data visible

    v16bf a[2], b[NT];
    #pragma unroll
    for (int mt = 0; mt < 2; ++mt) {        // A fragment (16x32, ISA 16-bit A layout)
      int m = wm * 32 + mt * 16 + l15;
      #pragma unroll
      for (int i = 0; i < 8; ++i) {
        a[mt][i]     = As[cur][m][hl * 8 + i];
        a[mt][i + 8] = As[cur][m][16 + hl * 8 + i];
      }
    }
    #pragma unroll
    for (int nt = 0; nt < NT; ++nt) {       // B fragment (32x16): n=l15, k=hl*8+i (i<8), 16+hl*8+(i-8) (i>=8)
      int n = wn * WN + nt * 16 + l15;
      #pragma unroll
      for (int i = 0; i < 16; ++i) b[nt][i] = Ws[cur][n][(i < 8 ? hl * 8 + i : 16 + hl * 8 + (i - 8))];
    }
    #pragma unroll
    for (int mt = 0; mt < 2; ++mt)
      #pragma unroll
      for (int nt = 0; nt < NT; ++nt)
        acc[mt][nt] = __builtin_amdgcn_wmma_f32_16x16x32_bf16(
            false, a[mt], false, b[nt], (short)0, acc[mt][nt], false, false);
    __syncthreads();   // compute(kc) done before anyone streams into buf[cur]
  }

  #pragma unroll
  for (int mt = 0; mt < 2; ++mt)
    #pragma unroll
    for (int nt = 0; nt < NT; ++nt)
      #pragma unroll
      for (int r = 0; r < 8; ++r) {
        int m = bm0 + wm * 32 + mt * 16 + r + hl * 8;
        int n = bn0 + wn * WN + nt * 16 + l15;
        C[(size_t)m * N + n] = acc[mt][nt][r];
      }
}

// ---------------------------------------------------------------- RMSNorm (f32 in -> bf16 out)
__global__ __launch_bounds__(256)
void mla_rmsnorm_bf16(const float* __restrict__ src, int sstride, int n,
                      const float* __restrict__ w,
                      bf16_t* __restrict__ dst, int dstride) {
  int row = blockIdx.x;
  const float* x = src + (size_t)row * sstride;
  float ss = 0.f;
  for (int i = threadIdx.x; i < n; i += blockDim.x) { float v = x[i]; ss += v * v; }
  #pragma unroll
  for (int m = 16; m >= 1; m >>= 1) ss += __shfl_xor(ss, m, 32);
  __shared__ float red[8];
  __shared__ float rms_s;
  int wv = threadIdx.x >> 5;
  if ((threadIdx.x & 31) == 0) red[wv] = ss;
  __syncthreads();
  if (threadIdx.x == 0) {
    float t = 0.f;
    for (int i = 0; i < (int)(blockDim.x >> 5); ++i) t += red[i];
    rms_s = rsqrtf(t / (float)n + 1e-6f);
  }
  __syncthreads();
  float rms = rms_s;
  for (int i = threadIdx.x; i < n; i += blockDim.x)
    dst[(size_t)row * dstride + i] = (bf16_t)(x[i] * rms * w[i]);
}

// ---------------------------------------------------------------- RoPE + q assembly
__global__ __launch_bounds__(256)
void mla_rope_assemble(const float* __restrict__ q_f, const float* __restrict__ kv_f,
                       bf16_t* __restrict__ q_b, float* __restrict__ krope_f) {
  int t = blockIdx.x;
  float pos = (float)t;
  const float lnTheta = 13.1223633774f; // log(500000)
  for (int idx = threadIdx.x; idx < H_HEADS * NOPE_; idx += blockDim.x) {
    int h = idx >> 7, d = idx & 127;
    q_b[((size_t)t * H_HEADS + h) * QKH_ + d] =
        (bf16_t)q_f[(size_t)t * H_HEADS * QKH_ + h * QKH_ + d];
  }
  for (int pi = threadIdx.x; pi < H_HEADS * (ROPE_ / 2); pi += blockDim.x) {
    int h = pi >> 5, i = pi & 31;
    float freq = __expf(-(2.f * i / (float)ROPE_) * lnTheta);
    float ang = pos * freq;
    float c = __cosf(ang), s = __sinf(ang);
    const float* qp = q_f + (size_t)t * H_HEADS * QKH_ + h * QKH_ + NOPE_ + 2 * i;
    float e = qp[0], o = qp[1];
    bf16_t* qo = q_b + ((size_t)t * H_HEADS + h) * QKH_ + NOPE_ + 2 * i;
    qo[0] = (bf16_t)(e * c - o * s);
    qo[1] = (bf16_t)(o * c + e * s);
  }
  if (threadIdx.x < ROPE_ / 2) {
    int i = threadIdx.x;
    float freq = __expf(-(2.f * i / (float)ROPE_) * lnTheta);
    float ang = pos * freq;
    float c = __cosf(ang), s = __sinf(ang);
    const float* kp = kv_f + (size_t)t * (KV_RANK_ + ROPE_) + KV_RANK_ + 2 * i;
    float e = kp[0], o = kp[1];
    krope_f[(size_t)t * ROPE_ + 2 * i]     = e * c - o * s;
    krope_f[(size_t)t * ROPE_ + 2 * i + 1] = o * c + e * s;
  }
}

// ---------------------------------------------------------------- split kv_up into k_full / v
__global__ __launch_bounds__(256)
void mla_split_kv_up(const float* __restrict__ kvup_f, const float* __restrict__ krope_f,
                     bf16_t* __restrict__ kfull_b, bf16_t* __restrict__ v_b) {
  int t = blockIdx.x;
  for (int idx = threadIdx.x; idx < H_HEADS * (NOPE_ + VD_); idx += blockDim.x) {
    int h = idx >> 8, d = idx & 255;
    float v = kvup_f[(size_t)t * H_HEADS * (NOPE_ + VD_) + h * (NOPE_ + VD_) + d];
    if (d < NOPE_) kfull_b[((size_t)t * H_HEADS + h) * QKH_ + d] = (bf16_t)v;
    else           v_b[((size_t)t * H_HEADS + h) * VD_ + (d - NOPE_)] = (bf16_t)v;
  }
  for (int idx = threadIdx.x; idx < H_HEADS * ROPE_; idx += blockDim.x) {
    int h = idx >> 6, i = idx & 63;
    kfull_b[((size_t)t * H_HEADS + h) * QKH_ + NOPE_ + i] =
        (bf16_t)krope_f[(size_t)t * ROPE_ + i];
  }
}

// ---------------------------------------------------------------- flash attention (causal)
#define AQT 64   // queries per block (4 waves x 16)
#define AKT 32   // keys per chunk
__global__ __launch_bounds__(128)
void mla_flash_attn(const bf16_t* __restrict__ Q, const bf16_t* __restrict__ Kf,
                    const bf16_t* __restrict__ V, float* __restrict__ O) {
  __shared__ bf16_t Ks[AKT][QKH_ + 8];    // K chunk, row major over d
  __shared__ bf16_t Vt[VD_][AKT + 8];     // V chunk transposed: [vd][key]
  __shared__ bf16_t Ps[4][16][AKT + 8];   // per-wave P tile
  const int h   = blockIdx.y;
  const int q0  = blockIdx.x * AQT;
  const int tid = threadIdx.x;
  const int lane = tid & 31, wv = tid >> 5;
  const int hl = lane >> 4, l15 = lane & 15;
  const int qw = q0 + wv * 16;

  v16bf qf[6];
  {
    const bf16_t* qp = Q + ((size_t)(qw + l15) * H_HEADS + h) * QKH_;
    #pragma unroll
    for (int c = 0; c < 6; ++c) {
      #pragma unroll
      for (int i = 0; i < 8; ++i) {
        qf[c][i]     = qp[c * 32 + hl * 8 + i];
        qf[c][i + 8] = qp[c * 32 + 16 + hl * 8 + i];
      }
    }
  }

  v8f acc[8];
  #pragma unroll
  for (int vt = 0; vt < 8; ++vt)
    #pragma unroll
    for (int i = 0; i < 8; ++i) acc[vt][i] = 0.f;
  float mrun[8], lrun[8];
  #pragma unroll
  for (int r = 0; r < 8; ++r) { mrun[r] = -1e30f; lrun[r] = 0.f; }

  const float scale = 0.07216878364870323f; // 1/sqrt(192)
  const int nchunk = (q0 + AQT) / AKT;

  for (int kc = 0; kc < nchunk; ++kc) {
    const int k0 = kc * AKT;
    {
      int r = tid >> 2, cb = (tid & 3) * 48;
      const bf16_t* g = Kf + ((size_t)(k0 + r) * H_HEADS + h) * QKH_ + cb;
      #pragma unroll
      for (int j = 0; j < 48; ++j) Ks[r][cb + j] = g[j];
    }
    {
      int kr = tid >> 2, vb = (tid & 3) * 32;
      const bf16_t* g = V + ((size_t)(k0 + kr) * H_HEADS + h) * VD_ + vb;
      #pragma unroll
      for (int j = 0; j < 32; ++j) Vt[vb + j][kr] = g[j];
    }
    __syncthreads();

    v8f s[2];
    #pragma unroll
    for (int nt = 0; nt < 2; ++nt) {
      #pragma unroll
      for (int i = 0; i < 8; ++i) s[nt][i] = 0.f;
      int key = nt * 16 + l15;
      #pragma unroll
      for (int c = 0; c < 6; ++c) {
        v16bf b;
        #pragma unroll
        for (int i = 0; i < 16; ++i) b[i] = Ks[key][c * 32 + (i < 8 ? hl * 8 + i : 16 + hl * 8 + (i - 8))];
        s[nt] = __builtin_amdgcn_wmma_f32_16x16x32_bf16(
            false, qf[c], false, b, (short)0, s[nt], false, false);
      }
      #pragma unroll
      for (int r = 0; r < 8; ++r) {
        int qrow = qw + r + hl * 8;
        int kidx = k0 + nt * 16 + l15;
        s[nt][r] = (kidx <= qrow) ? s[nt][r] * scale : -1e30f;
      }
    }

    float alpha[8];
    #pragma unroll
    for (int r = 0; r < 8; ++r) {
      float rm = fmaxf(s[0][r], s[1][r]);
      #pragma unroll
      for (int msk = 1; msk < 16; msk <<= 1) rm = fmaxf(rm, __shfl_xor(rm, msk, 32));
      float mnew = fmaxf(mrun[r], rm);
      alpha[r] = __expf(mrun[r] - mnew);
      float p0 = __expf(s[0][r] - mnew);
      float p1 = __expf(s[1][r] - mnew);
      s[0][r] = p0; s[1][r] = p1;
      float ls = p0 + p1;
      #pragma unroll
      for (int msk = 1; msk < 16; msk <<= 1) ls += __shfl_xor(ls, msk, 32);
      lrun[r] = lrun[r] * alpha[r] + ls;
      mrun[r] = mnew;
    }
    #pragma unroll
    for (int vt = 0; vt < 8; ++vt)
      #pragma unroll
      for (int r = 0; r < 8; ++r) acc[vt][r] *= alpha[r];

    #pragma unroll
    for (int r = 0; r < 8; ++r) {
      int m = r + hl * 8;
      Ps[wv][m][l15]      = (bf16_t)s[0][r];
      Ps[wv][m][16 + l15] = (bf16_t)s[1][r];
    }
    asm volatile("s_wait_dscnt 0" ::: "memory");
    v16bf pf;
    #pragma unroll
    for (int i = 0; i < 8; ++i) {
      pf[i]     = Ps[wv][l15][hl * 8 + i];
      pf[i + 8] = Ps[wv][l15][16 + hl * 8 + i];
    }

    #pragma unroll
    for (int vt = 0; vt < 8; ++vt) {
      v16bf b;
      int vd = vt * 16 + l15;
      #pragma unroll
      for (int i = 0; i < 16; ++i) b[i] = Vt[vd][(i < 8 ? hl * 8 + i : 16 + hl * 8 + (i - 8))];
      acc[vt] = __builtin_amdgcn_wmma_f32_16x16x32_bf16(
          false, pf, false, b, (short)0, acc[vt], false, false);
    }
    __syncthreads();
  }

  #pragma unroll
  for (int r = 0; r < 8; ++r) {
    int qrow = qw + r + hl * 8;
    float inv = 1.f / lrun[r];
    #pragma unroll
    for (int vt = 0; vt < 8; ++vt)
      O[(size_t)qrow * (H_HEADS * VD_) + h * VD_ + vt * 16 + l15] = acc[vt][r] * inv;
  }
}

// ---------------------------------------------------------------- host launcher
static inline size_t mla_align256(size_t x) { return (x + 255) & ~(size_t)255; }

extern "C" void kernel_launch(void* const* d_in, const int* in_sizes, int n_in,
                              void* d_out, int out_size, void* d_ws, size_t ws_size,
                              hipStream_t stream) {
  (void)in_sizes; (void)n_in; (void)out_size; (void)ws_size;
  const float* x      = (const float*)d_in[0];
  const float* wqd    = (const float*)d_in[1];
  const float* qnw    = (const float*)d_in[2];
  const float* wqu    = (const float*)d_in[3];
  const float* wkvd   = (const float*)d_in[4];
  const float* kvnw   = (const float*)d_in[5];
  const float* wkvu   = (const float*)d_in[6];
  const float* wo     = (const float*)d_in[7];
  float* out = (float*)d_out;

  char* w = (char*)d_ws;
  auto alloc = [&](size_t bytes) { void* p = (void*)w; w += mla_align256(bytes); return p; };

  const size_t nX    = (size_t)T_SEQ * DIM_;
  const size_t nWqd  = (size_t)Q_RANK_ * DIM_;
  const size_t nWqu  = (size_t)H_HEADS * QKH_ * Q_RANK_;
  const size_t nWkvd = (size_t)(KV_RANK_ + ROPE_) * DIM_;
  const size_t nWkvu = (size_t)H_HEADS * (NOPE_ + VD_) * KV_RANK_;
  const size_t nWo   = (size_t)DIM_ * H_HEADS * VD_;

  bf16_t* xb     = (bf16_t*)alloc(nX * 2);
  bf16_t* wqd_b  = (bf16_t*)alloc(nWqd * 2);
  bf16_t* wqu_b  = (bf16_t*)alloc(nWqu * 2);
  bf16_t* wkvd_b = (bf16_t*)alloc(nWkvd * 2);
  bf16_t* wkvu_b = (bf16_t*)alloc(nWkvu * 2);
  bf16_t* wo_b   = (bf16_t*)alloc(nWo * 2);
  float*  cq_f   = (float*) alloc((size_t)T_SEQ * Q_RANK_ * 4);
  bf16_t* cq_b   = (bf16_t*)alloc((size_t)T_SEQ * Q_RANK_ * 2);
  float*  q_f    = (float*) alloc((size_t)T_SEQ * H_HEADS * QKH_ * 4);
  bf16_t* q_b    = (bf16_t*)alloc((size_t)T_SEQ * H_HEADS * QKH_ * 2);
  float*  kv_f   = (float*) alloc((size_t)T_SEQ * (KV_RANK_ + ROPE_) * 4);
  bf16_t* ckv_b  = (bf16_t*)alloc((size_t)T_SEQ * KV_RANK_ * 2);
  float*  kvup_f = (float*) alloc((size_t)T_SEQ * H_HEADS * (NOPE_ + VD_) * 4);
  float*  krope_f= (float*) alloc((size_t)T_SEQ * ROPE_ * 4);
  bf16_t* kfull_b= (bf16_t*)alloc((size_t)T_SEQ * H_HEADS * QKH_ * 2);
  bf16_t* v_b    = (bf16_t*)alloc((size_t)T_SEQ * H_HEADS * VD_ * 2);
  float*  ao_f   = (float*) alloc((size_t)T_SEQ * H_HEADS * VD_ * 4);
  bf16_t* ao_b   = (bf16_t*)alloc((size_t)T_SEQ * H_HEADS * VD_ * 2);

  auto cast = [&](const float* s, bf16_t* d, size_t n) {
    mla_cast_f32_bf16<<<(unsigned)((n + 255) / 256), 256, 0, stream>>>(s, d, (int)n);
  };
  cast(x, xb, nX);
  cast(wqd, wqd_b, nWqd);
  cast(wqu, wqu_b, nWqu);
  cast(wkvd, wkvd_b, nWkvd);
  cast(wkvu, wkvu_b, nWkvu);
  cast(wo, wo_b, nWo);

  // c_q = x @ Wqd^T  [2048,1536]
  mla_gemm_bf16_wmma<128><<<dim3(Q_RANK_ / 128, T_SEQ / 128), 256, 0, stream>>>(
      xb, wqd_b, cq_f, T_SEQ, Q_RANK_, DIM_);
  // kv = x @ Wkvd^T  [2048,576]  (576 % 128 != 0 -> 64-wide tiles)
  mla_gemm_bf16_wmma<64><<<dim3((KV_RANK_ + ROPE_) / 64, T_SEQ / 128), 256, 0, stream>>>(
      xb, wkvd_b, kv_f, T_SEQ, KV_RANK_ + ROPE_, DIM_);

  mla_rmsnorm_bf16<<<T_SEQ, 256, 0, stream>>>(cq_f, Q_RANK_, Q_RANK_, qnw, cq_b, Q_RANK_);
  mla_rmsnorm_bf16<<<T_SEQ, 256, 0, stream>>>(kv_f, KV_RANK_ + ROPE_, KV_RANK_, kvnw, ckv_b, KV_RANK_);

  // q = c_q @ Wqu^T  [2048,3072]
  mla_gemm_bf16_wmma<128><<<dim3((H_HEADS * QKH_) / 128, T_SEQ / 128), 256, 0, stream>>>(
      cq_b, wqu_b, q_f, T_SEQ, H_HEADS * QKH_, Q_RANK_);
  // kv_up = ckv @ Wkvu^T  [2048,4096]
  mla_gemm_bf16_wmma<128><<<dim3((H_HEADS * (NOPE_ + VD_)) / 128, T_SEQ / 128), 256, 0, stream>>>(
      ckv_b, wkvu_b, kvup_f, T_SEQ, H_HEADS * (NOPE_ + VD_), KV_RANK_);

  mla_rope_assemble<<<T_SEQ, 256, 0, stream>>>(q_f, kv_f, q_b, krope_f);
  mla_split_kv_up<<<T_SEQ, 256, 0, stream>>>(kvup_f, krope_f, kfull_b, v_b);

  mla_flash_attn<<<dim3(T_SEQ / AQT, H_HEADS), 128, 0, stream>>>(q_b, kfull_b, v_b, ao_f);

  cast(ao_f, ao_b, (size_t)T_SEQ * H_HEADS * VD_);

  // out = attn_out @ Wo^T  [2048,2048]
  mla_gemm_bf16_wmma<128><<<dim3(DIM_ / 128, T_SEQ / 128), 256, 0, stream>>>(
      ao_b, wo_b, out, T_SEQ, DIM_, H_HEADS * VD_);
}
